// Attn_8701603741852
// MI455X (gfx1250) — matrix operand order repaired and checked by reading, not compiled, not run on hardware
//
#include <hip/hip_runtime.h>
#include <hip/hip_bf16.h>

typedef __attribute__((ext_vector_type(16))) __bf16 v16bf;
typedef __attribute__((ext_vector_type(8)))  float  v8f;

#define TT   4096
#define HH   512
#define BB   32
#define ROWS 64          // rows of x per block in the scores kernel
#define LSTRIDE 520      // padded LDS row stride in bf16 elems (1040B = 260 dwords -> +4 banks/row)
#define NSPLIT 16        // T-splits for the weighted-sum pass

union FragAB { uint4 q[2]; v16bf v; };

static __device__ __forceinline__ unsigned short f2bf(float f) {
    unsigned u = __float_as_uint(f);
    unsigned r = (u + 0x7FFFu + ((u >> 16) & 1u)) >> 16;   // round-to-nearest-even
    return (unsigned short)r;
}

// async copy of 16 bytes global -> LDS (GLOBAL_LOAD_ASYNC_TO_LDS_B128, ASYNCcnt-tracked)
static __device__ __forceinline__ void async_copy_b128(unsigned ldsByteOff, const void* gaddr) {
    asm volatile("global_load_async_to_lds_b128 %0, %1, off"
                 :: "v"(ldsByteOff), "v"((unsigned long long)(uintptr_t)gaddr)
                 : "memory");
}
static __device__ __forceinline__ void wait_async0() {
    asm volatile("s_wait_asynccnt 0x0" ::: "memory");
}

// ---------------------------------------------------------------- W -> bf16
__global__ __launch_bounds__(256) void attn_convW(const float* __restrict__ W,
                                                  unsigned short* __restrict__ wbf) {
    int i = blockIdx.x * 256 + threadIdx.x;
    if (i < HH * HH) wbf[i] = f2bf(W[i]);
}

// ------------------------------------------- fused tanh(x@W^T) . v  -> scores
__global__ __launch_bounds__(128) void attn_scores(const float* __restrict__ x,
                                                   const unsigned short* __restrict__ wbf,
                                                   const float* __restrict__ v,
                                                   float* __restrict__ scores) {
    extern __shared__ unsigned short smem[];
    unsigned short* Xt  = smem;                           // ROWS x LSTRIDE (bf16)
    unsigned short* Wt0 = smem + ROWS * LSTRIDE;          // 16 x LSTRIDE, buffer 0
    unsigned short* Wt1 = Wt0 + 16 * LSTRIDE;             // 16 x LSTRIDE, buffer 1

    const int  tid     = threadIdx.x;
    const long rowBase = (long)blockIdx.x * ROWS;

    // LDS byte offsets of the two W buffers (for async-copy VDST)
    const unsigned wtOff0 = (unsigned)(uintptr_t)Wt0;
    const unsigned wtOff1 = (unsigned)(uintptr_t)Wt1;

    // ---- stage x tile (f32 -> bf16) into LDS (needs conversion -> VALU path)
    const float4* xsrc = (const float4*)(x + rowBase * HH);
    for (int i = tid; i < ROWS * HH / 4; i += 128) {
        float4 f = xsrc[i];
        int e = i * 4;
        int r = e >> 9;          // / 512
        int c = e & 511;
        unsigned short* p = Xt + r * LSTRIDE + c;
        p[0] = f2bf(f.x); p[1] = f2bf(f.y); p[2] = f2bf(f.z); p[3] = f2bf(f.w);
    }

    // ---- kick off async copy of W tile 0 (16 x 512 bf16 = 1024 x 16B; 8 per thread)
#pragma unroll
    for (int it = 0; it < 8; ++it) {
        int i = tid + it * 128;
        int e = i * 8;
        int r = e >> 9;
        int c = e & 511;
        async_copy_b128(wtOff0 + (unsigned)(r * LSTRIDE + c) * 2,
                        wbf + (size_t)r * HH + c);
    }
    __syncthreads();   // X tile visible

    const int lane = tid & 31;
    const int wid  = tid >> 5;
    const int lm   = lane & 15;
    const int half = lane >> 4;
    const int r0   = wid * 16;   // this wave's 16 rows within the tile

    // ---- A fragments for all 16 K-steps, kept in VGPRs
    // A(16x32 bf16): lane holds row lm; lo lanes K = k0+0..7,16..23 ; hi lanes +8
    FragAB a[16];
#pragma unroll
    for (int ks = 0; ks < 16; ++ks) {
        const unsigned short* base = Xt + (r0 + lm) * LSTRIDE + ks * 32 + half * 8;
        a[ks].q[0] = *(const uint4*)(base);        // K offsets 0..7
        a[ks].q[1] = *(const uint4*)(base + 16);   // K offsets 16..23
    }

    float sacc[8];
#pragma unroll
    for (int j = 0; j < 8; ++j) sacc[j] = 0.0f;

    for (int nc = 0; nc < 32; ++nc) {               // 32 N-chunks of 16 columns
        const int cur = nc & 1;
        wait_async0();        // W tile nc resident in buf[cur]
        __syncthreads();      // all waves past wait; all waves done computing tile nc-1

        // stream W tile nc+1 into the other buffer while we compute on this one
        if (nc + 1 < 32) {
            const unsigned dstOff = (cur ? wtOff0 : wtOff1);
#pragma unroll
            for (int it = 0; it < 8; ++it) {
                int i = tid + it * 128;
                int e = i * 8;
                int r = e >> 9;
                int c = e & 511;
                async_copy_b128(dstOff + (unsigned)(r * LSTRIDE + c) * 2,
                                wbf + (size_t)((nc + 1) * 16 + r) * HH + c);
            }
        }

        // ---- 16 K-steps of WMMA with pipelined B-fragment loads + dual accumulators
        const unsigned short* wt = cur ? Wt1 : Wt0;
        const unsigned short* bbase = wt + lm * LSTRIDE + half * 8;
        // B(32x16 bf16): lane holds column lm; lo lanes K = 0..7,16..23 ; hi lanes +8
        FragAB b[2];
        b[0].q[0] = *(const uint4*)(bbase);
        b[0].q[1] = *(const uint4*)(bbase + 16);
        v8f acc0 = {}, acc1 = {};
#pragma unroll
        for (int ks = 0; ks < 16; ++ks) {
            if (ks + 1 < 16) {
                const unsigned short* nb = bbase + (ks + 1) * 32;
                b[(ks + 1) & 1].q[0] = *(const uint4*)(nb);
                b[(ks + 1) & 1].q[1] = *(const uint4*)(nb + 16);
            }
            if (ks & 1)
                acc1 = __builtin_amdgcn_wmma_f32_16x16x32_bf16(
                    false, a[ks].v, false, b[ks & 1].v, (short)0, acc1, false, false);
            else
                acc0 = __builtin_amdgcn_wmma_f32_16x16x32_bf16(
                    false, a[ks].v, false, b[ks & 1].v, (short)0, acc0, false, false);
        }
        v8f acc = acc0 + acc1;

        // epilogue: tanh * v[n], accumulate per lane (lane lm covers N % 16 == lm)
        const float vv = v[nc * 16 + lm];
#pragma unroll
        for (int j = 0; j < 8; ++j) sacc[j] += tanhf(acc[j]) * vv;
    }

    // cross-lane reduce within each 16-lane half; C-layout: VGPR j, lo lanes M=j, hi lanes M=j+8
#pragma unroll
    for (int j = 0; j < 8; ++j) {
        float s = sacc[j];
        s += __shfl_xor(s, 1, 32);
        s += __shfl_xor(s, 2, 32);
        s += __shfl_xor(s, 4, 32);
        s += __shfl_xor(s, 8, 32);
        if (lm == 0) scores[rowBase + r0 + half * 8 + j] = s;
    }
}

// ---------------------------------------------------------- softmax over T
__global__ __launch_bounds__(256) void attn_softmax(float* __restrict__ scores) {
    const int b = blockIdx.x;
    float* s = scores + (size_t)b * TT;
    const int tid  = threadIdx.x;
    const int lane = tid & 31;
    const int wid  = tid >> 5;
    __shared__ float red[8];

    float vals[16];
    float m = -3.402823466e+38f;
#pragma unroll
    for (int i = 0; i < 16; ++i) { vals[i] = s[tid + i * 256]; m = fmaxf(m, vals[i]); }
    m = fmaxf(m, __shfl_xor(m, 1, 32));
    m = fmaxf(m, __shfl_xor(m, 2, 32));
    m = fmaxf(m, __shfl_xor(m, 4, 32));
    m = fmaxf(m, __shfl_xor(m, 8, 32));
    m = fmaxf(m, __shfl_xor(m, 16, 32));
    if (lane == 0) red[wid] = m;
    __syncthreads();
    float bm = red[0];
#pragma unroll
    for (int i = 1; i < 8; ++i) bm = fmaxf(bm, red[i]);
    __syncthreads();

    float sum = 0.0f;
#pragma unroll
    for (int i = 0; i < 16; ++i) { vals[i] = __expf(vals[i] - bm); sum += vals[i]; }
    sum += __shfl_xor(sum, 1, 32);
    sum += __shfl_xor(sum, 2, 32);
    sum += __shfl_xor(sum, 4, 32);
    sum += __shfl_xor(sum, 8, 32);
    sum += __shfl_xor(sum, 16, 32);
    if (lane == 0) red[wid] = sum;
    __syncthreads();
    float tot = 0.0f;
#pragma unroll
    for (int i = 0; i < 8; ++i) tot += red[i];
    const float inv = 1.0f / tot;
#pragma unroll
    for (int i = 0; i < 16; ++i) s[tid + i * 256] = vals[i] * inv;
}

// ------------------------------------- out partials: sum_t w[t] * x[t,:] per T-split
__global__ __launch_bounds__(256) void attn_wsum_partial(const float* __restrict__ x,
                                                         const float* __restrict__ w,
                                                         float* __restrict__ partials) {
    const int ts  = blockIdx.x;   // 0..NSPLIT-1
    const int b   = blockIdx.y;   // 0..BB-1
    const int tid = threadIdx.x;
    const int tchunk = TT / NSPLIT;               // 256
    const long base  = ((long)b * TT + (long)ts * tchunk) * HH;
    const float* wp  = w + (size_t)b * TT + (size_t)ts * tchunk;

    float a0 = 0.0f, a1 = 0.0f;
    for (int t = 0; t < tchunk; ++t) {
        const float wt = wp[t];
        a0 += wt * x[base + (long)t * HH + tid];
        a1 += wt * x[base + (long)t * HH + tid + 256];
    }
    float* p = partials + ((size_t)ts * BB + b) * HH;
    p[tid]       = a0;
    p[tid + 256] = a1;
}

__global__ __launch_bounds__(256) void attn_wsum_reduce(const float* __restrict__ partials,
                                                        float* __restrict__ out) {
    const int i = blockIdx.x * 256 + threadIdx.x;   // 0 .. BB*HH-1
    float s = 0.0f;
#pragma unroll
    for (int ts = 0; ts < NSPLIT; ++ts) s += partials[(size_t)ts * BB * HH + i];
    out[i] = s;
}

// ---------------------------------------------------------------------------
extern "C" void kernel_launch(void* const* d_in, const int* in_sizes, int n_in,
                              void* d_out, int out_size, void* d_ws, size_t ws_size,
                              hipStream_t stream) {
    const float* x = (const float*)d_in[0];   // (32,4096,512)
    const float* W = (const float*)d_in[1];   // (512,512)
    const float* v = (const float*)d_in[2];   // (512,)
    float* out = (float*)d_out;               // (32,512)

    // workspace layout
    unsigned short* wbf   = (unsigned short*)d_ws;                       // 512*512*2  = 512 KB
    float* scores         = (float*)((char*)d_ws + 524288);              // 32*4096*4  = 512 KB
    float* partials       = (float*)((char*)d_ws + 1048576);             // 16*32*512*4 = 1 MB

    attn_convW<<<(HH * HH + 255) / 256, 256, 0, stream>>>(W, wbf);

    const int nblk = (BB * TT) / ROWS;                                   // 2048
    const size_t shmem = (size_t)(ROWS + 32) * LSTRIDE * sizeof(unsigned short); // ~100 KB
    attn_scores<<<nblk, 128, shmem, stream>>>(x, wbf, v, scores);

    attn_softmax<<<BB, 256, 0, stream>>>(scores);

    attn_wsum_partial<<<dim3(NSPLIT, BB), 256, 0, stream>>>(x, scores, partials);

    attn_wsum_reduce<<<(BB * HH) / 256, 256, 0, stream>>>(partials, out);
}
